// mIHC_hypergraph_surv_83494164234286
// MI455X (gfx1250) — matrix operand order repaired and checked by reading, not compiled, not run on hardware
//
#include <hip/hip_runtime.h>

#define NNODES 200000
#define EHYP   25000
#define NNZ_   1600000
#define NG     16
#define CIN    64
#define HD     128
#define BN_EPS 1e-5f

typedef __attribute__((ext_vector_type(16))) __bf16 v16bf;
typedef __attribute__((ext_vector_type(8)))  float  v8f;

union Frag16 {
  v16bf v;
  uint4 q[2];
  unsigned short h[16];
};

static __device__ __forceinline__ unsigned short f2bf(float f) {
  unsigned int u = __float_as_uint(f);
  u += 0x7FFFu + ((u >> 16) & 1u);   // round-to-nearest-even
  return (unsigned short)(u >> 16);
}

// ---------------------------------------------------------------------------
// WMMA GEMM over node features: O[M,128] = act(A[M,KDIM] @ W[KDIM,128] (+bias))
// One wave per 16-row M tile; weights pre-packed as bf16 B fragments in LDS.
// B layout (16-bit, 32x16): lanes 0-15: K 0-7 / 16-23; lanes 16-31: K 8-15 / 24-31.
// A layout (16-bit, 16x32): lanes 0-15: K 0-7 / 16-23; lanes 16-31: K 8-15 / 24-31.
// ---------------------------------------------------------------------------
template<int KDIM, bool A_F32, bool OUT_BF16_RELU>
__global__ __launch_bounds__(256) void gemm_node_k(
    const void* __restrict__ Aptr, const float* __restrict__ W,
    const float* __restrict__ bias, void* __restrict__ Optr, int mtiles)
{
  constexpr int KT = KDIM / 32;
  __shared__ unsigned short Bf[KT * 8 * 32 * 16];

  const int tid = threadIdx.x;
  // Fill fragment-ready bf16 weights into LDS.
  for (int i = tid; i < KT * 8 * 32 * 16; i += 256) {
    int e = i & 15;
    int lane = (i >> 4) & 31;
    int f = i >> 9;
    int ktile = f >> 3, ntile = f & 7;
    int k = ktile * 32 + (lane >> 4) * 8 + (e & 7) + (e >> 3) * 16;
    int n = ntile * 16 + (lane & 15);
    Bf[i] = f2bf(W[k * HD + n]);
  }
  __syncthreads();

  const int lane = tid & 31;
  const int wave = tid >> 5;
  const int m = lane & 15;
  const int kh = lane >> 4;

  for (int tile = blockIdx.x * 8 + wave; tile < mtiles; tile += gridDim.x * 8) {
    Frag16 a[KT];
    const size_t rowbase = (size_t)(tile * 16 + m) * KDIM;
    if constexpr (A_F32) {
      const float* Ar = (const float*)Aptr + rowbase;
#pragma unroll
      for (int kt = 0; kt < KT; ++kt) {
        const int c0 = kt * 32 + kh * 8;
        float4 f0 = *(const float4*)(Ar + c0);
        float4 f1 = *(const float4*)(Ar + c0 + 4);
        float4 f2 = *(const float4*)(Ar + c0 + 16);
        float4 f3 = *(const float4*)(Ar + c0 + 20);
        a[kt].h[0]  = f2bf(f0.x); a[kt].h[1]  = f2bf(f0.y);
        a[kt].h[2]  = f2bf(f0.z); a[kt].h[3]  = f2bf(f0.w);
        a[kt].h[4]  = f2bf(f1.x); a[kt].h[5]  = f2bf(f1.y);
        a[kt].h[6]  = f2bf(f1.z); a[kt].h[7]  = f2bf(f1.w);
        a[kt].h[8]  = f2bf(f2.x); a[kt].h[9]  = f2bf(f2.y);
        a[kt].h[10] = f2bf(f2.z); a[kt].h[11] = f2bf(f2.w);
        a[kt].h[12] = f2bf(f3.x); a[kt].h[13] = f2bf(f3.y);
        a[kt].h[14] = f2bf(f3.z); a[kt].h[15] = f2bf(f3.w);
      }
    } else {
      const unsigned short* Ar = (const unsigned short*)Aptr + rowbase;
#pragma unroll
      for (int kt = 0; kt < KT; ++kt) {
        const int c0 = kt * 32 + kh * 8;
        a[kt].q[0] = *(const uint4*)(Ar + c0);
        a[kt].q[1] = *(const uint4*)(Ar + c0 + 16);
      }
    }

    v8f acc[8];
#pragma unroll
    for (int nt = 0; nt < 8; ++nt) acc[nt] = {};

#pragma unroll
    for (int kt = 0; kt < KT; ++kt) {
#pragma unroll
      for (int nt = 0; nt < 8; ++nt) {
        Frag16 b;
        const uint4* bp = (const uint4*)&Bf[(((kt * 8 + nt) * 32) + lane) * 16];
        b.q[0] = bp[0];
        b.q[1] = bp[1];
        acc[nt] = __builtin_amdgcn_wmma_f32_16x16x32_bf16(
            false, a[kt].v, false, b.v, (short)0, acc[nt], false, false);
      }
    }

    // C/D layout: VGPR r -> row (r + 8*kh), col = nt*16 + (lane&15)
    const int colb = lane & 15;
    const size_t orow0 = (size_t)(tile * 16 + kh * 8) * HD;
#pragma unroll
    for (int nt = 0; nt < 8; ++nt) {
#pragma unroll
      for (int r = 0; r < 8; ++r) {
        float v = acc[nt][r];
        const size_t oidx = orow0 + (size_t)r * HD + nt * 16 + colb;
        if constexpr (OUT_BF16_RELU) {
          v = fmaxf(v + bias[nt * 16 + colb], 0.0f);
          ((unsigned short*)Optr)[oidx] = f2bf(v);
        } else {
          ((float*)Optr)[oidx] = v;
        }
      }
    }
  }
}

// ---------------------------------------------------------------------------
__global__ __launch_bounds__(256) void zero_f32_k(float* __restrict__ p, int n) {
  int i = blockIdx.x * 256 + threadIdx.x;
  if (i < n) p[i] = 0.0f;
}

__global__ __launch_bounds__(256) void degrees_k(
    const int* __restrict__ row, const int* __restrict__ col,
    float* __restrict__ dn, float* __restrict__ de) {
  int i = blockIdx.x * 256 + threadIdx.x;
  if (i < NNZ_) {
    atomicAdd(&dn[row[i]], 1.0f);
    atomicAdd(&de[col[i]], 1.0f);
  }
}

__global__ __launch_bounds__(256) void counts_k(
    const int* __restrict__ batch, float* __restrict__ cnt) {
  int i = blockIdx.x * 256 + threadIdx.x;
  if (i < NNODES) atomicAdd(&cnt[batch[i]], 1.0f);
}

// node -> hyperedge scatter: he[col] += X[row]   (32 lanes = one full row)
__global__ __launch_bounds__(256) void scatter_n2e_k(
    const float* __restrict__ X, const int* __restrict__ row,
    const int* __restrict__ col, float* __restrict__ he) {
  int t = blockIdx.x * 256 + threadIdx.x;
  int e = t >> 5;
  if (e >= NNZ_) return;
  int c4 = (t & 31) << 2;
  int r = row[e], c = col[e];
  const float4 v = *(const float4*)(X + (size_t)r * HD + c4);
  float* dst = he + (size_t)c * HD + c4;
  atomicAdd(dst + 0, v.x);
  atomicAdd(dst + 1, v.y);
  atomicAdd(dst + 2, v.z);
  atomicAdd(dst + 3, v.w);
}

// hyperedge -> node scatter with B^{-1} scaling: Y[row] += he[col] / deg_e[col]
__global__ __launch_bounds__(256) void scatter_e2n_k(
    const float* __restrict__ he, const float* __restrict__ de,
    const int* __restrict__ row, const int* __restrict__ col,
    float* __restrict__ Y) {
  int t = blockIdx.x * 256 + threadIdx.x;
  int e = t >> 5;
  if (e >= NNZ_) return;
  int c4 = (t & 31) << 2;
  int r = row[e], c = col[e];
  float d = de[c];
  float s = d > 0.0f ? 1.0f / d : 0.0f;
  const float4 v = *(const float4*)(he + (size_t)c * HD + c4);
  float* dst = Y + (size_t)r * HD + c4;
  atomicAdd(dst + 0, v.x * s);
  atomicAdd(dst + 1, v.y * s);
  atomicAdd(dst + 2, v.z * s);
  atomicAdd(dst + 3, v.w * s);
}

// Fused: h = relu(D^{-1}*Y + b); write bf16 for next GEMM; accumulate mean-pool
// sums (batch is sorted -> local accumulation, few atomics).
template<bool WRITE_BF16>
__global__ __launch_bounds__(128) void finalize_k(
    const float* __restrict__ Y, const float* __restrict__ dn,
    const int* __restrict__ batch, const float* __restrict__ bias,
    unsigned short* __restrict__ hA, float* __restrict__ pool) {
  const int col = threadIdx.x;            // 0..127
  const int base = blockIdx.x * 64;       // 64 nodes per block
  const float b = bias[col];
  float acc = 0.0f;
  int curg = -1;
  for (int n = 0; n < 64; ++n) {
    const int node = base + n;
    const float d = dn[node];
    const float dinv = d > 0.0f ? 1.0f / d : 0.0f;
    float val = fmaxf(Y[(size_t)node * HD + col] * dinv + b, 0.0f);
    if constexpr (WRITE_BF16) hA[(size_t)node * HD + col] = f2bf(val);
    const int g = batch[node];
    if (g != curg) {
      if (curg >= 0) atomicAdd(&pool[curg * HD + col], acc);
      curg = g;
      acc = 0.0f;
    }
    acc += val;
  }
  if (curg >= 0) atomicAdd(&pool[curg * HD + col], acc);
}

// ---------------------------------------------------------------------------
// Whole MLP head in one block: pools -> Wg -> (W1,BN,relu) -> (W2,BN,relu) -> Wout
// ---------------------------------------------------------------------------
__global__ __launch_bounds__(256) void final_mlp_k(
    const float* __restrict__ pool1, const float* __restrict__ pool2,
    const float* __restrict__ cnt,
    const float* __restrict__ Wg, const float* __restrict__ bg,
    const float* __restrict__ W1, const float* __restrict__ b1,
    const float* __restrict__ g1, const float* __restrict__ be1,
    const float* __restrict__ W2, const float* __restrict__ b2,
    const float* __restrict__ g2, const float* __restrict__ be2,
    const float* __restrict__ Wout, float* __restrict__ out)
{
  __shared__ float P[NG * 256];
  __shared__ float G[NG * 128];
  __shared__ float T1[NG * 64];
  __shared__ float T2[NG * 32];
  const int tid = threadIdx.x;

  for (int i = tid; i < NG * 256; i += 256) {
    int g = i >> 8, c = i & 255;
    float cc = fmaxf(cnt[g], 1.0f);
    float s = (c < 128) ? pool1[g * 128 + c] : pool2[g * 128 + (c - 128)];
    P[i] = s / cc;
  }
  __syncthreads();

  for (int i = tid; i < NG * 128; i += 256) {
    int g = i >> 7, c = i & 127;
    float s = bg[c];
    for (int k = 0; k < 256; ++k) s += P[g * 256 + k] * Wg[k * 128 + c];
    G[i] = s;
  }
  __syncthreads();

  for (int i = tid; i < NG * 64; i += 256) {
    int g = i >> 6, c = i & 63;
    float s = b1[c];
    for (int k = 0; k < 128; ++k) s += G[g * 128 + k] * W1[k * 64 + c];
    T1[i] = s;
  }
  __syncthreads();

  if (tid < 64) {
    float m = 0.0f;
    for (int g = 0; g < NG; ++g) m += T1[g * 64 + tid];
    m *= (1.0f / NG);
    float v = 0.0f;
    for (int g = 0; g < NG; ++g) { float d = T1[g * 64 + tid] - m; v += d * d; }
    v *= (1.0f / NG);
    float inv = g1[tid] / sqrtf(v + BN_EPS);
    for (int g = 0; g < NG; ++g)
      T1[g * 64 + tid] = fmaxf((T1[g * 64 + tid] - m) * inv + be1[tid], 0.0f);
  }
  __syncthreads();

  for (int i = tid; i < NG * 32; i += 256) {
    int g = i >> 5, c = i & 31;
    float s = b2[c];
    for (int k = 0; k < 64; ++k) s += T1[g * 64 + k] * W2[k * 32 + c];
    T2[i] = s;
  }
  __syncthreads();

  if (tid < 32) {
    float m = 0.0f;
    for (int g = 0; g < NG; ++g) m += T2[g * 32 + tid];
    m *= (1.0f / NG);
    float v = 0.0f;
    for (int g = 0; g < NG; ++g) { float d = T2[g * 32 + tid] - m; v += d * d; }
    v *= (1.0f / NG);
    float inv = g2[tid] / sqrtf(v + BN_EPS);
    for (int g = 0; g < NG; ++g)
      T2[g * 32 + tid] = fmaxf((T2[g * 32 + tid] - m) * inv + be2[tid], 0.0f);
  }
  __syncthreads();

  if (tid < NG * 4) {
    int g = tid >> 2, c = tid & 3;
    float s = 0.0f;
    for (int k = 0; k < 32; ++k) s += T2[g * 32 + k] * Wout[k * 4 + c];
    out[tid] = s;
  }
}

// ---------------------------------------------------------------------------
extern "C" void kernel_launch(void* const* d_in, const int* in_sizes, int n_in,
                              void* d_out, int out_size, void* d_ws, size_t ws_size,
                              hipStream_t stream) {
  (void)in_sizes; (void)n_in; (void)out_size; (void)ws_size;
  const float* x    = (const float*)d_in[0];
  const int*   erow = (const int*)d_in[1];
  const int*   ecol = (const int*)d_in[2];
  const int*   batch= (const int*)d_in[3];
  const float* W0   = (const float*)d_in[4];
  const float* b0   = (const float*)d_in[5];
  const float* Wc1  = (const float*)d_in[6];
  const float* bc1  = (const float*)d_in[7];
  const float* Wc2  = (const float*)d_in[8];
  const float* bc2  = (const float*)d_in[9];
  const float* Wg   = (const float*)d_in[10];
  const float* bg   = (const float*)d_in[11];
  const float* W1   = (const float*)d_in[12];
  const float* b1   = (const float*)d_in[13];
  const float* g1   = (const float*)d_in[14];
  const float* be1  = (const float*)d_in[15];
  const float* W2   = (const float*)d_in[16];
  const float* b2   = (const float*)d_in[17];
  const float* g2   = (const float*)d_in[18];
  const float* be2  = (const float*)d_in[19];
  const float* Wout = (const float*)d_in[20];
  float* out = (float*)d_out;

  char* w = (char*)d_ws;
  size_t off = 0;
  auto take = [&](size_t bytes) {
    char* p = w + off;
    off = (off + bytes + 255) & ~(size_t)255;
    return p;
  };
  unsigned short* hA = (unsigned short*)take((size_t)NNODES * HD * 2); // bf16 acts
  float* X     = (float*)take((size_t)NNODES * HD * 4);  // GEMM out / scatter target
  float* he    = (float*)take((size_t)EHYP * HD * 4);
  float* dn    = (float*)take((size_t)NNODES * 4);
  float* de    = (float*)take((size_t)EHYP * 4);
  float* cnt   = (float*)take(NG * 4);
  float* pool1 = (float*)take(NG * HD * 4);
  float* pool2 = (float*)take(NG * HD * 4);

  auto blocks = [](long long n) { return dim3((unsigned)((n + 255) / 256)); };

  // Degrees / counts / pool init
  zero_f32_k<<<blocks(NNODES), 256, 0, stream>>>(dn, NNODES);
  zero_f32_k<<<blocks(EHYP),   256, 0, stream>>>(de, EHYP);
  zero_f32_k<<<1,              256, 0, stream>>>(cnt, NG);
  zero_f32_k<<<blocks(NG*HD),  256, 0, stream>>>(pool1, NG * HD);
  zero_f32_k<<<blocks(NG*HD),  256, 0, stream>>>(pool2, NG * HD);
  degrees_k<<<blocks(NNZ_),    256, 0, stream>>>(erow, ecol, dn, de);
  counts_k<<<blocks(NNODES),   256, 0, stream>>>(batch, cnt);

  const int mtiles = NNODES / 16;               // 12500
  const int gblocks = (mtiles + 7) / 8;

  // h0 = relu(x @ W0 + b0) -> bf16
  gemm_node_k<CIN, true, true><<<gblocks, 256, 0, stream>>>(x, W0, b0, hA, mtiles);

  const long long scat = (long long)NNZ_ * 32;

  // hconv1
  gemm_node_k<HD, false, false><<<gblocks, 256, 0, stream>>>(hA, Wc1, nullptr, X, mtiles);
  zero_f32_k<<<blocks((long long)EHYP * HD), 256, 0, stream>>>(he, EHYP * HD);
  scatter_n2e_k<<<blocks(scat), 256, 0, stream>>>(X, erow, ecol, he);
  zero_f32_k<<<blocks((long long)NNODES * HD), 256, 0, stream>>>(X, NNODES * HD);
  scatter_e2n_k<<<blocks(scat), 256, 0, stream>>>(he, de, erow, ecol, X);
  finalize_k<true><<<NNODES / 64, 128, 0, stream>>>(X, dn, batch, bc1, hA, pool1);

  // hconv2
  gemm_node_k<HD, false, false><<<gblocks, 256, 0, stream>>>(hA, Wc2, nullptr, X, mtiles);
  zero_f32_k<<<blocks((long long)EHYP * HD), 256, 0, stream>>>(he, EHYP * HD);
  scatter_n2e_k<<<blocks(scat), 256, 0, stream>>>(X, erow, ecol, he);
  zero_f32_k<<<blocks((long long)NNODES * HD), 256, 0, stream>>>(X, NNODES * HD);
  scatter_e2n_k<<<blocks(scat), 256, 0, stream>>>(he, de, erow, ecol, X);
  finalize_k<false><<<NNODES / 64, 128, 0, stream>>>(X, dn, batch, bc2, nullptr, pool2);

  // Head
  final_mlp_k<<<1, 256, 0, stream>>>(pool1, pool2, cnt, Wg, bg,
                                     W1, b1, g1, be1, W2, b2, g2, be2, Wout, out);
}
